// AlexNetOWT_BN_71562745086149
// MI455X (gfx1250) — 8-bit matrix operand order repaired and checked by reading, compiled, not run on hardware
//
#include <hip/hip_runtime.h>
#include <stdint.h>

typedef int v8i __attribute__((ext_vector_type(8)));
typedef int v4i __attribute__((ext_vector_type(4)));
typedef __attribute__((address_space(1))) v4i* gptr4;
typedef __attribute__((address_space(3))) v4i* lptr4;

#define BN_EPS 1e-5f

#if defined(__has_builtin)
#if __has_builtin(__builtin_amdgcn_global_load_async_to_lds_b128)
#define HAVE_ASYNC_LDS 1
#endif
#endif
#ifndef HAVE_ASYNC_LDS
#define HAVE_ASYNC_LDS 0
#endif

__device__ __forceinline__ void wait_async0() {
#if __has_builtin(__builtin_amdgcn_s_wait_asynccnt)
  __builtin_amdgcn_s_wait_asynccnt(0);
#else
  asm volatile("s_wait_asynccnt 0" ::: "memory");
#endif
}

__device__ __forceinline__ v8i v8i_zero() {
  v8i v;
#pragma unroll
  for (int i = 0; i < 8; ++i) v[i] = 0;
  return v;
}

__device__ __forceinline__ signed char sgnf(float x) {
  return (signed char)((x > 0.f) - (x < 0.f));
}

__global__ void prepack_weights(const float* __restrict__ w_mid,
                                const float* __restrict__ w_last,
                                int* __restrict__ midpack,
                                int* __restrict__ lastpack) {
  int idx = blockIdx.x * blockDim.x + threadIdx.x;
  const int NMID = 5 * 2 * 21 * 32;
  if (idx < NMID) {
    int lane = idx & 31;
    int u = idx >> 5;
    int it = u % 21;
    int w2 = u / 21;
    int g = w2 & 1;
    int layer = w2 >> 1;
    int hi = lane >> 4;
    int m = lane & 15;
    int cout = g * 16 + m;
#pragma unroll
    for (int v = 0; v < 8; ++v) {
      int q = v & 3, h = v >> 2;
      int k0 = (h << 5) + ((q >> 1) << 4) + ((q & 1) << 2) + (hi << 3);
      unsigned dw = 0;
#pragma unroll
      for (int bb = 0; bb < 4; ++bb) {
        int k = k0 + bb;
        int k2 = k >> 5, ci = k & 31;
        int tap = 2 * it + k2;
        signed char sv = 0;
        if (tap <= 40) {
          float w = w_mid[(((layer * 32 + cout) * 32 + ci) * 41) + tap];
          sv = sgnf(w);
        }
        dw |= ((unsigned)(unsigned char)sv) << (8 * bb);
      }
      midpack[idx * 8 + v] = (int)dw;
    }
  } else if (idx < NMID + 21 * 32) {
    int j = idx - NMID;
    int lane = j & 31;
    int it = j >> 5;
    int hi = lane >> 4;
    int cout = lane & 15;
#pragma unroll
    for (int v = 0; v < 8; ++v) {
      int q = v & 3, h = v >> 2;
      int k0 = (h << 5) + ((q >> 1) << 4) + ((q & 1) << 2) + (hi << 3);
      unsigned dw = 0;
#pragma unroll
      for (int bb = 0; bb < 4; ++bb) {
        int k = k0 + bb;
        int k2 = k >> 5, ci = k & 31;
        int tap = 2 * it + k2;
        signed char sv = 0;
        if (tap <= 40) {
          float w = w_last[((cout * 32 + ci) * 41) + tap];
          sv = sgnf(w);
        }
        dw |= ((unsigned)(unsigned char)sv) << (8 * bb);
      }
      lastpack[j * 8 + v] = (int)dw;
    }
  }
}

__global__ void first_conv_bn_sign(const float* __restrict__ x,
                                   const float* __restrict__ w_first,
                                   const float* __restrict__ b_first,
                                   const float* __restrict__ bng,
                                   const float* __restrict__ bnb,
                                   const float* __restrict__ bnm,
                                   const float* __restrict__ bnv,
                                   int8_t* __restrict__ actOut) {
  __shared__ float sw[32 * 3 * 41];
  __shared__ float sx[3 * 168];
  const int tid = threadIdx.x;
  const int b = blockIdx.y;
  const int l0 = blockIdx.x * 128;
  for (int i = tid; i < 32 * 3 * 41; i += 128) sw[i] = (float)sgnf(w_first[i]);
  for (int i = tid; i < 3 * 168; i += 128) {
    int ci = i / 168, t = i - ci * 168;
    int gl = l0 + t;
    float v = (gl < 16000) ? x[(b * 3 + ci) * 16000 + gl] : 0.f;
    sx[i] = fminf(1.f, fmaxf(-1.f, v));
  }
  __syncthreads();
  int l = l0 + tid;
  if (l >= 15960) return;
  __align__(16) signed char obuf[32];
  for (int co = 0; co < 32; ++co) {
    float acc = 0.f;
#pragma unroll
    for (int ci = 0; ci < 3; ++ci) {
      const float* xs = &sx[ci * 168 + tid];
      const float* ws = &sw[(co * 3 + ci) * 41];
      for (int k = 0; k < 41; ++k) acc += xs[k] * ws[k];
    }
    float f = acc + b_first[co];
    float t = bng[co] * (f - bnm[co]) * rsqrtf(bnv[co] + BN_EPS) + bnb[co];
    obuf[co] = sgnf(t);
  }
  int4* dst = (int4*)(actOut + ((size_t)b * 15960 + l) * 32);
  dst[0] = ((const int4*)obuf)[0];
  dst[1] = ((const int4*)obuf)[1];
}

template <int DIL, int NG, bool RAW>
__global__ void bconv_wmma(const int8_t* __restrict__ actIn, int Lin, int Lout,
                           const int* __restrict__ wpack,
                           const float* __restrict__ bias,
                           const float* __restrict__ bng,
                           const float* __restrict__ bnb,
                           const float* __restrict__ bnm,
                           const float* __restrict__ bnv,
                           int8_t* __restrict__ actOut,
                           float* __restrict__ rawOut) {
  constexpr int RALLOC = 128 + 40 * DIL;
  extern __shared__ char smem[];
  const int tid = threadIdx.x;
  const int lane = tid & 31, wid = tid >> 5;
  const int b = blockIdx.y;
  const int l0 = blockIdx.x * 128;

  const int8_t* gbase = actIn + ((size_t)b * Lin + l0) * 32;
#pragma unroll 1
  for (int c = tid; c < 2 * RALLOC; c += 256) {
    int off = c * 16;
    int l = l0 + (off >> 5);
    if (l < Lin) {
#if HAVE_ASYNC_LDS
      __builtin_amdgcn_global_load_async_to_lds_b128(
          (gptr4)(gbase + off), (lptr4)(smem + off), 0, 0);
#else
      *(int4*)(smem + off) = *(const int4*)(gbase + off);
#endif
    } else {
      int4 z;
      z.x = z.y = z.z = z.w = 0;
      *(int4*)(smem + off) = z;
    }
  }
#if HAVE_ASYNC_LDS
  wait_async0();
#endif
  __syncthreads();

  const int hi = lane >> 4, nn = lane & 15;
  const int nbase = wid * 16;
  const int kgrp = hi << 4;
  v8i acc0 = v8i_zero(), acc1 = v8i_zero();
  const v8i* wp = (const v8i*)wpack;

#pragma unroll
  for (int it = 0; it < 21; ++it) {
    v8i bf;
#pragma unroll
    for (int v = 0; v < 8; ++v) {
      int kbase = ((v >> 2) << 5) + (((v & 3) >> 1) << 4) + ((v & 1) << 2) + (hi << 3);
      int k2 = kbase >> 5;
      int tap = 2 * it + k2;
      int ci0 = kbase & 31;
      int lloc = nbase + nn + tap * DIL;
      bf[v] = (tap <= 40) ? *(const int*)(smem + lloc * 32 + ci0) : 0;
    }
    v8i af0 = wp[it * 32 + lane];
    acc0 = __builtin_amdgcn_wmma_i32_16x16x64_iu8(true, af0, true, bf, acc0,
                                                  false, false);
    if (NG > 1) {
      v8i af1 = wp[(21 + it) * 32 + lane];
      acc1 = __builtin_amdgcn_wmma_i32_16x16x64_iu8(true, af1, true, bf, acc1,
                                                    false, false);
    }
  }

  int l = l0 + nbase + nn;
  if (l >= Lout) return;
#pragma unroll
  for (int g = 0; g < NG; ++g) {
    v8i a = (g == 0) ? acc0 : acc1;
    if constexpr (RAW) {
#pragma unroll
      for (int r = 0; r < 8; ++r) {
        int co = g * 16 + hi * 8 + r;
        rawOut[((size_t)(b * 16 + co)) * Lout + l] = (float)a[r] + bias[co];
      }
    } else {
      uint64_t pk = 0;
#pragma unroll
      for (int r = 0; r < 8; ++r) {
        int co = g * 16 + hi * 8 + r;
        float f = (float)a[r] + bias[co];
        float t = bng[co] * (f - bnm[co]) * rsqrtf(bnv[co] + BN_EPS) + bnb[co];
        pk |= ((uint64_t)(uint8_t)sgnf(t)) << (8 * r);
      }
      *(uint64_t*)(actOut + ((size_t)b * Lout + l) * 32 + g * 16 + hi * 8) = pk;
    }
  }
}

__global__ void pool4_sign(const float* __restrict__ conv,
                           int8_t* __restrict__ afc) {
  int idx = blockIdx.x * blockDim.x + threadIdx.x;
  const int KP = 43712;
  if (idx >= 32 * KP) return;
  int b = idx / KP, kf = idx - b * KP;
  signed char v = 0;
  if (kf < 43680) {
    int c = kf / 2730, lp = kf - c * 2730;
    const float* p = conv + ((b * 16 + c) * 10920) + 4 * lp;
    float m = fmaxf(fmaxf(p[0], p[1]), fmaxf(p[2], p[3]));
    v = sgnf(m);
  }
  afc[idx] = v;
}

__global__ void fc_wmma(const int8_t* __restrict__ afc,
                        const float* __restrict__ fc_w,
                        const float* __restrict__ fc_b,
                        float* __restrict__ zbuf) {
  const int lane = threadIdx.x & 31;
  const int ntile = blockIdx.x, mt = blockIdx.y;
  const int hi = lane >> 4, nn = lane & 15;
  const int kgrp = hi << 4;
  const int ncls = ntile * 16 + nn;
  const int mrow = mt * 16 + nn;
  const bool nval = (ncls < 1000);
  const int8_t* arow = afc + (size_t)mrow * 43712;
  const float* wrow = fc_w + (size_t)(nval ? ncls : 0) * 43680;

  v8i acc = v8i_zero();
  for (int kit = 0; kit < 683; ++kit) {
    if (nval) __builtin_prefetch(wrow + kit * 64 + 2048, 0, 0);
    v8i af, bf;
#pragma unroll
    for (int v = 0; v < 8; ++v) {
      int q = v & 3, h = v >> 2;
      int k0 = (h << 5) + ((q >> 1) << 4) + ((q & 1) << 2) + (hi << 3);
      af[v] = *(const int*)(arow + kit * 64 + k0);
      int kbase = k0;
      int kk = kit * 64 + kbase;
      unsigned dw = 0;
      if (nval) {
#pragma unroll
        for (int bb = 0; bb < 4; ++bb) {
          int kz = kk + bb;
          signed char sv = 0;
          if (kz < 43680) sv = sgnf(wrow[kz]);
          dw |= ((unsigned)(unsigned char)sv) << (8 * bb);
        }
      }
      bf[v] = (int)dw;
    }
    acc = __builtin_amdgcn_wmma_i32_16x16x64_iu8(true, af, true, bf, acc,
                                                 false, false);
  }
  if (nval) {
#pragma unroll
    for (int r = 0; r < 8; ++r) {
      int m = mt * 16 + hi * 8 + r;
      zbuf[m * 1000 + ncls] = (float)acc[r] + fc_b[ncls];
    }
  }
}

__global__ void bn_logsoftmax(const float* __restrict__ z,
                              const float* __restrict__ cg,
                              const float* __restrict__ cb,
                              const float* __restrict__ cm,
                              const float* __restrict__ cv,
                              float* __restrict__ out) {
  __shared__ float red[256];
  const int b = blockIdx.x, tid = threadIdx.x;
  float lmax = -1e30f;
  for (int n = tid; n < 1000; n += 256) {
    float t = cg[n] * (z[b * 1000 + n] - cm[n]) * rsqrtf(cv[n] + BN_EPS) + cb[n];
    out[b * 1000 + n] = t;
    lmax = fmaxf(lmax, t);
  }
  red[tid] = lmax;
  __syncthreads();
  for (int s = 128; s > 0; s >>= 1) {
    if (tid < s) red[tid] = fmaxf(red[tid], red[tid + s]);
    __syncthreads();
  }
  float mx = red[0];
  __syncthreads();
  float lsum = 0.f;
  for (int n = tid; n < 1000; n += 256) lsum += __expf(out[b * 1000 + n] - mx);
  red[tid] = lsum;
  __syncthreads();
  for (int s = 128; s > 0; s >>= 1) {
    if (tid < s) red[tid] += red[tid + s];
    __syncthreads();
  }
  float lse = logf(red[0]) + mx;
  __syncthreads();
  for (int n = tid; n < 1000; n += 256) out[b * 1000 + n] -= lse;
}

extern "C" void kernel_launch(void* const* d_in, const int* in_sizes, int n_in,
                              void* d_out, int out_size, void* d_ws,
                              size_t ws_size, hipStream_t stream) {
  (void)in_sizes; (void)n_in; (void)out_size; (void)ws_size;
  const float* x       = (const float*)d_in[0];
  const float* w_first = (const float*)d_in[1];
  const float* b_first = (const float*)d_in[2];
  const float* w_mid   = (const float*)d_in[3];
  const float* b_mid   = (const float*)d_in[4];
  const float* w_last  = (const float*)d_in[5];
  const float* b_last  = (const float*)d_in[6];
  const float* bn_g    = (const float*)d_in[7];
  const float* bn_b    = (const float*)d_in[8];
  const float* bn_m    = (const float*)d_in[9];
  const float* bn_v    = (const float*)d_in[10];
  const float* fc_w    = (const float*)d_in[11];
  const float* fc_b    = (const float*)d_in[12];
  const float* cg      = (const float*)d_in[13];
  const float* cb      = (const float*)d_in[14];
  const float* cm      = (const float*)d_in[15];
  const float* cv      = (const float*)d_in[16];
  float* out = (float*)d_out;

  char* ws = (char*)d_ws;
  size_t off = 0;
  auto walloc = [&](size_t bytes) -> char* {
    char* p = ws + off;
    off += (bytes + 255) & ~(size_t)255;
    return p;
  };
  int8_t* actA     = (int8_t*)walloc((size_t)32 * 32 * 15960);
  int8_t* actB     = (int8_t*)walloc((size_t)32 * 32 * 15960);
  int*    midpack  = (int*)walloc((size_t)5 * 2 * 21 * 32 * 32);
  int*    lastpack = (int*)walloc((size_t)21 * 32 * 32);
  float*  convout  = (float*)walloc(sizeof(float) * 32 * 16 * 10920);
  int8_t* afc      = (int8_t*)walloc((size_t)32 * 43712);
  float*  zbuf     = (float*)walloc(sizeof(float) * 32 * 1000);

  prepack_weights<<<dim3((7392 + 255) / 256), dim3(256), 0, stream>>>(
      w_mid, w_last, midpack, lastpack);

  first_conv_bn_sign<<<dim3(125, 32), dim3(128), 0, stream>>>(
      x, w_first, b_first, bn_g, bn_b, bn_m, bn_v, actA);

#define LAUNCH_MID(I, DILV, LINV, LOUTV, PIN, POUT)                           \
  bconv_wmma<DILV, 2, false><<<dim3((LOUTV + 127) / 128, 32), dim3(256),      \
                               (size_t)32 * (128 + 40 * DILV), stream>>>(     \
      PIN, LINV, LOUTV, midpack + (I) * (2 * 21 * 32 * 8), b_mid + (I) * 32,  \
      bn_g + ((I) + 1) * 32, bn_b + ((I) + 1) * 32, bn_m + ((I) + 1) * 32,    \
      bn_v + ((I) + 1) * 32, POUT, nullptr)

  LAUNCH_MID(0, 2, 15960, 15880, actA, actB);
  LAUNCH_MID(1, 4, 15880, 15720, actB, actA);
  LAUNCH_MID(2, 8, 15720, 15400, actA, actB);
  LAUNCH_MID(3, 16, 15400, 14760, actB, actA);
  LAUNCH_MID(4, 32, 14760, 13480, actA, actB);
#undef LAUNCH_MID

  bconv_wmma<64, 1, true><<<dim3((10920 + 127) / 128, 32), dim3(256),
                            (size_t)32 * (128 + 40 * 64), stream>>>(
      actB, 13480, 10920, lastpack, b_last, nullptr, nullptr, nullptr, nullptr,
      nullptr, convout);

  pool4_sign<<<dim3((32 * 43712 + 255) / 256), dim3(256), 0, stream>>>(convout,
                                                                       afc);
  fc_wmma<<<dim3(63, 2), dim3(32), 0, stream>>>(afc, fc_w, fc_b, zbuf);
  bn_logsoftmax<<<dim3(32), dim3(256), 0, stream>>>(zbuf, cg, cb, cm, cv, out);
}
